// _PointnetSAModuleBase_86517821215288
// MI455X (gfx1250) — compile-verified, matrix operand order checked by reading, not run on hardware
//
#include <hip/hip_runtime.h>

typedef __attribute__((ext_vector_type(16))) _Float16 v16h;
typedef __attribute__((ext_vector_type(8)))  float    v8f;

#define NPTS    8192
#define NBATCH  8
#define NPOINT  1024
#define NSAMPLE 32
#define CFEAT   64
#define K0PAD   96
#define RAD2    0.04f

// ---------------------------------------------------------------------------
// Kernel 1: fold scale into weights, convert to f16, zero-pad layer0 K to 96.
// ---------------------------------------------------------------------------
__global__ void prep_weights(const float* __restrict__ w0, const float* __restrict__ s0,
                             const float* __restrict__ w1, const float* __restrict__ s1,
                             const float* __restrict__ w2, const float* __restrict__ s2,
                             _Float16* __restrict__ w0h, _Float16* __restrict__ w1h,
                             _Float16* __restrict__ w2h) {
  int t = blockIdx.x * 256 + threadIdx.x;
  if (t < 64 * K0PAD) {                       // w0h: 64 x 96  (from 64 x 67)
    int o = t / K0PAD, c = t % K0PAD;
    float v = (c < 67) ? w0[o * 67 + c] * s0[o] : 0.f;
    w0h[t] = (_Float16)v;
  }
  if (t < 64 * 64) {                          // w1h: 64 x 64
    int o = t >> 6;
    w1h[t] = (_Float16)(w1[t] * s1[o]);
  }
  if (t < 128 * 64) {                         // w2h: 128 x 64
    int o = t >> 6;
    w2h[t] = (_Float16)(w2[t] * s2[o]);
  }
}

// ---------------------------------------------------------------------------
// Kernel 2: furthest point sampling. One block per batch, all 8192 points
// held in registers (16 per thread), 1024 sequential argmax rounds with
// wave32 shuffle reduction + tiny LDS cross-wave step. Writes new_xyz.
// ---------------------------------------------------------------------------
__global__ __launch_bounds__(512) void fps_kernel(const float* __restrict__ xyz,
                                                  float* __restrict__ newxyz) {
  const int b   = blockIdx.x;
  const int tid = threadIdx.x;
  const float* xb = xyz + (size_t)b * NPTS * 3;

  float px[16], py[16], pz[16], mind[16];
#pragma unroll
  for (int k = 0; k < 16; ++k) {
    int p = tid + (k << 9);
    px[k] = xb[p * 3 + 0];
    py[k] = xb[p * 3 + 1];
    pz[k] = xb[p * 3 + 2];
    mind[k] = 1e10f;
  }

  __shared__ float sLast[3];
  __shared__ float sD[16];
  __shared__ int   sI[16];
  __shared__ int   sWin;

  if (tid == 0) {                 // i = 0 -> index 0 always (reference semantics)
    sLast[0] = px[0]; sLast[1] = py[0]; sLast[2] = pz[0];
    newxyz[(b * NPOINT + 0) * 3 + 0] = px[0];
    newxyz[(b * NPOINT + 0) * 3 + 1] = py[0];
    newxyz[(b * NPOINT + 0) * 3 + 2] = pz[0];
  }
  __syncthreads();

  const int lane = tid & 31;
  const int wav  = tid >> 5;

  for (int i = 1; i < NPOINT; ++i) {
    float lx = sLast[0], ly = sLast[1], lz = sLast[2];
    float bd = -1.f; int bi = 0;
#pragma unroll
    for (int k = 0; k < 16; ++k) {
      float dx = px[k] - lx, dy = py[k] - ly, dz = pz[k] - lz;
      float d = dx * dx + dy * dy + dz * dz;
      float m = fminf(mind[k], d);
      mind[k] = m;
      int p = tid + (k << 9);
      if (m > bd || (m == bd && p < bi)) { bd = m; bi = p; }
    }
    // wave32 argmax (first-index tie break, matches jnp.argmax)
#pragma unroll
    for (int off = 16; off > 0; off >>= 1) {
      float od = __shfl_xor(bd, off, 32);
      int   oi = __shfl_xor(bi, off, 32);
      if (od > bd || (od == bd && oi < bi)) { bd = od; bi = oi; }
    }
    if (lane == 0) { sD[wav] = bd; sI[wav] = bi; }
    __syncthreads();
    if (wav == 0) {
      float d2 = (lane < 16) ? sD[lane] : -1.f;
      int   i2 = (lane < 16) ? sI[lane] : 0;
#pragma unroll
      for (int off = 16; off > 0; off >>= 1) {
        float od = __shfl_xor(d2, off, 32);
        int   oi = __shfl_xor(i2, off, 32);
        if (od > d2 || (od == d2 && oi < i2)) { d2 = od; i2 = oi; }
      }
      if (lane == 0) sWin = i2;
    }
    __syncthreads();
    int nxt = sWin;
    // owning thread publishes winner coords from registers (no HBM round trip)
#pragma unroll
    for (int k = 0; k < 16; ++k) {
      if (tid + (k << 9) == nxt) {
        sLast[0] = px[k]; sLast[1] = py[k]; sLast[2] = pz[k];
        newxyz[(b * NPOINT + i) * 3 + 0] = px[k];
        newxyz[(b * NPOINT + i) * 3 + 1] = py[k];
        newxyz[(b * NPOINT + i) * 3 + 2] = pz[k];
      }
    }
    __syncthreads();
  }
}

// ---------------------------------------------------------------------------
// WMMA fragment loader: 16 rows x 32 K slice, f16, row stride in halves.
// Per lane: rows = lane&15, K chunks [kb,kb+8) and [kb+16,kb+24), kb=(lane>>4)*8
// -> exactly the CDNA5 16-bit A/B VGPR layout; two 16-byte loads.
// ---------------------------------------------------------------------------
__device__ inline v16h load_frag(const _Float16* __restrict__ p0, int lane) {
  const _Float16* p = p0 + ((lane >> 4) << 3);
  union { v16h v; float4 f[2]; } u;
  u.f[0] = *(const float4*)(p);
  u.f[1] = *(const float4*)(p + 16);
  return u.v;
}

// ---------------------------------------------------------------------------
// Kernel 3: one block (4 waves) per group: ball query (ordered first-32
// selection with ballot + early exit), gather into LDS (f16), 3-layer MLP
// with v_wmma_f32_16x16x32_f16, ReLU+bias, column max, write (B,128,S).
// ---------------------------------------------------------------------------
__global__ __launch_bounds__(128) void group_mlp_kernel(
    const float* __restrict__ xyz, const float* __restrict__ feat,
    const float* __restrict__ newxyz,
    const _Float16* __restrict__ w0h, const float* __restrict__ bias0,
    const _Float16* __restrict__ w1h, const float* __restrict__ bias1,
    const _Float16* __restrict__ w2h, const float* __restrict__ bias2,
    float* __restrict__ outFeat) {

  const int g    = blockIdx.x;        // b * 1024 + s
  const int b    = g >> 10;
  const int s    = g & 1023;
  const int tid  = threadIdx.x;
  const int lane = tid & 31;
  const int wav  = tid >> 5;

  __shared__ alignas(16) _Float16 X0[32 * K0PAD];
  __shared__ alignas(16) _Float16 H1[32 * 64];
  __shared__ alignas(16) _Float16 H2[32 * 64];
  __shared__ int   sIdx[NSAMPLE];
  __shared__ int   sCnt[4];
  __shared__ float sCtr[3];

  if (tid < 3)  sCtr[tid] = newxyz[(size_t)g * 3 + tid];
  if (tid < NSAMPLE) sIdx[tid] = 0;
  __syncthreads();
  const float cx = sCtr[0], cy = sCtr[1], cz = sCtr[2];
  const float* xb = xyz + (size_t)b * NPTS * 3;

  // ---- ball query: first 32 indices (ascending) with d2 < r2 ----
  int cnt = 0;
  for (int base = 0; base < NPTS && cnt < NSAMPLE; base += 128) {
    int p = base + tid;
    float dx = xb[p * 3 + 0] - cx;
    float dy = xb[p * 3 + 1] - cy;
    float dz = xb[p * 3 + 2] - cz;
    bool in = (dx * dx + dy * dy + dz * dz) < RAD2;
    unsigned long long m = __ballot(in);
    int wc = __popcll(m);
    if (lane == 0) sCnt[wav] = wc;
    __syncthreads();
    int pre = 0, tot = 0;
#pragma unroll
    for (int w = 0; w < 4; ++w) {
      int c = sCnt[w];
      if (w < wav) pre += c;
      tot += c;
    }
    int below = __popcll(m & ((1ull << lane) - 1ull));
    int rank = cnt + pre + below;
    if (in && rank < NSAMPLE) sIdx[rank] = p;
    cnt += tot;                      // uniform across block
    __syncthreads();
  }
  int first = (cnt > 0) ? sIdx[0] : 0;
  __syncthreads();
  if (tid < NSAMPLE && tid >= cnt) sIdx[tid] = first;   // fill semantics
  __syncthreads();

  // ---- gather: X0 row r = [xyz-rel(3) | feat(64) | pad->96], f16 ----
  {
    const int r = tid >> 2, q = tid & 3;
    int idx = sIdx[r];
    const float* fr = feat + ((size_t)b * NPTS + idx) * CFEAT + q * 16;
#pragma unroll
    for (int j = 0; j < 16; ++j)
      X0[r * K0PAD + 3 + q * 16 + j] = (_Float16)fr[j];
    if (q == 0) {
      X0[r * K0PAD + 0] = (_Float16)(xb[idx * 3 + 0] - cx);
      X0[r * K0PAD + 1] = (_Float16)(xb[idx * 3 + 1] - cy);
      X0[r * K0PAD + 2] = (_Float16)(xb[idx * 3 + 2] - cz);
    }
    for (int c = 67 + q * 8; c < K0PAD && c < 75 + q * 8; ++c)
      X0[r * K0PAD + c] = (_Float16)0.f;
  }
  __syncthreads();

  const int col = lane & 15;
  const int rh  = (lane >> 4) << 3;

  // ---- layer 0: (32x96)x(96x64) -> H1, wave w owns n-tile w ----
  {
    int nt = wav;
    float bv = bias0[nt * 16 + col];
#pragma unroll
    for (int mt = 0; mt < 2; ++mt) {
      v8f acc = {};
#pragma unroll
      for (int ks = 0; ks < 3; ++ks) {
        v16h a  = load_frag(X0  + (mt * 16 + col) * K0PAD + ks * 32, lane);
        v16h bm = load_frag(w0h + (nt * 16 + col) * K0PAD + ks * 32, lane);
        acc = __builtin_amdgcn_wmma_f32_16x16x32_f16(false, a, false, bm,
                                                     (short)0, acc, false, false);
      }
#pragma unroll
      for (int j = 0; j < 8; ++j) {
        float y = fmaxf(acc[j] + bv, 0.f);
        H1[(mt * 16 + rh + j) * 64 + nt * 16 + col] = (_Float16)y;
      }
    }
  }
  __syncthreads();

  // ---- layer 1: (32x64)x(64x64) -> H2 ----
  {
    int nt = wav;
    float bv = bias1[nt * 16 + col];
#pragma unroll
    for (int mt = 0; mt < 2; ++mt) {
      v8f acc = {};
#pragma unroll
      for (int ks = 0; ks < 2; ++ks) {
        v16h a  = load_frag(H1  + (mt * 16 + col) * 64 + ks * 32, lane);
        v16h bm = load_frag(w1h + (nt * 16 + col) * 64 + ks * 32, lane);
        acc = __builtin_amdgcn_wmma_f32_16x16x32_f16(false, a, false, bm,
                                                     (short)0, acc, false, false);
      }
#pragma unroll
      for (int j = 0; j < 8; ++j) {
        float y = fmaxf(acc[j] + bv, 0.f);
        H2[(mt * 16 + rh + j) * 64 + nt * 16 + col] = (_Float16)y;
      }
    }
  }
  __syncthreads();

  // ---- layer 2: (32x64)x(64x128) -> relu -> max over 32 rows -> out ----
#pragma unroll
  for (int nn = 0; nn < 2; ++nn) {
    int nt = wav + nn * 4;
    float bv = bias2[nt * 16 + col];
    float cm = 0.f;                  // ReLU output >= 0
#pragma unroll
    for (int mt = 0; mt < 2; ++mt) {
      v8f acc = {};
#pragma unroll
      for (int ks = 0; ks < 2; ++ks) {
        v16h a  = load_frag(H2  + (mt * 16 + col) * 64 + ks * 32, lane);
        v16h bm = load_frag(w2h + (nt * 16 + col) * 64 + ks * 32, lane);
        acc = __builtin_amdgcn_wmma_f32_16x16x32_f16(false, a, false, bm,
                                                     (short)0, acc, false, false);
      }
#pragma unroll
      for (int j = 0; j < 8; ++j)
        cm = fmaxf(cm, fmaxf(acc[j] + bv, 0.f));
    }
    // lane l and l+16 hold the same column (rows 0-7 / 8-15): one xor-16 max
    cm = fmaxf(cm, __shfl_xor(cm, 16, 32));
    if (lane < 16)
      outFeat[((size_t)b * 128 + nt * 16 + lane) * NPOINT + s] = cm;
  }
}

// ---------------------------------------------------------------------------
extern "C" void kernel_launch(void* const* d_in, const int* in_sizes, int n_in,
                              void* d_out, int out_size, void* d_ws, size_t ws_size,
                              hipStream_t stream) {
  (void)in_sizes; (void)n_in; (void)out_size; (void)ws_size;
  const float* xyz  = (const float*)d_in[0];
  const float* feat = (const float*)d_in[1];
  const float* w0 = (const float*)d_in[2];
  const float* s0 = (const float*)d_in[3];
  const float* b0 = (const float*)d_in[4];
  const float* w1 = (const float*)d_in[5];
  const float* s1 = (const float*)d_in[6];
  const float* b1 = (const float*)d_in[7];
  const float* w2 = (const float*)d_in[8];
  const float* s2 = (const float*)d_in[9];
  const float* b2 = (const float*)d_in[10];

  float* out     = (float*)d_out;
  float* newxyz  = out;                          // (8,1024,3)
  float* outFeat = out + NBATCH * NPOINT * 3;    // (8,128,1024)

  char* ws = (char*)d_ws;
  _Float16* w0h = (_Float16*)(ws);               // 64*96*2  = 12288 B
  _Float16* w1h = (_Float16*)(ws + 12288);       // 64*64*2  =  8192 B
  _Float16* w2h = (_Float16*)(ws + 20480);       // 128*64*2 = 16384 B

  prep_weights<<<32, 256, 0, stream>>>(w0, s0, w1, s1, w2, s2, w0h, w1h, w2h);
  fps_kernel<<<NBATCH, 512, 0, stream>>>(xyz, newxyz);
  group_mlp_kernel<<<NBATCH * NPOINT, 128, 0, stream>>>(
      xyz, feat, newxyz, w0h, b0, w1h, b1, w2h, b2, outFeat);
}
